// Encoder_34668976013746
// MI455X (gfx1250) — matrix operand order repaired and checked by reading, not compiled, not run on hardware
//
#include <hip/hip_runtime.h>
#include <hip/hip_bf16.h>
#include <math.h>

// ---------------------------------------------------------------------------
// MI455X (gfx1250) encoder implementation.
// Compute-bound (~365 GFLOP vs <0.5GB traffic @23.3TB/s) -> all big GEMMs go
// through V_WMMA_F32_16X16X32_F16 with f32->f16 conversion staged in LDS.
// Block tile 128x128, 8 waves, wave tile 32x64 = 8 wmma per K=32 step.
// FULL template = branch-free hot loop for the exactly-tileable GEMMs
// (all the FLOP-dominant ones); guarded path only for K=624 / M=32 cases.
// Scratch requirement: ~233 MB of d_ws.
// ---------------------------------------------------------------------------

typedef __attribute__((ext_vector_type(16))) _Float16 v16h;
typedef __attribute__((ext_vector_type(8)))  float    v8f;

#define BT_TOTAL (32 * 384)     // 12288 rows
#define DIMM 896
#define HH 128
#define HD 7
#define TT 384
#define BB 32

#define BM 128
#define BN 128
#define BK 32

// ---------------------------------------------------------------------------
// WMMA GEMM: C[M,N] = A[M,K] @ W[K,N] + bias   (A,W,C fp32; math in f16 WMMA)
// 8 waves: wm = wave&3 (4 x 32 rows), wn = wave>>2 (2 x 64 cols).
// N must be a multiple of 128. FULL => M%128==0 && K%32==0.
// ---------------------------------------------------------------------------
template <bool FULL>
__global__ __launch_bounds__(256)
void wmma_gemm_kernel(const float* __restrict__ A, const float* __restrict__ W,
                      const float* __restrict__ bias, float* __restrict__ C,
                      int M, int N, int K)
{
    __shared__ _Float16 lA[BM * BK];   // row-major [m][k]
    __shared__ _Float16 lB[BN * BK];   // transposed [n][k]

    const int tid  = threadIdx.x;
    const int lane = tid & 31;
    const int wave = tid >> 5;      // 0..7
    const int wm   = wave & 3;      // 4 waves along M (32 rows each)
    const int wn   = wave >> 2;     // 2 waves along N (64 cols each)
    const int m0   = blockIdx.y * BM;
    const int n0   = blockIdx.x * BN;
    const int half = lane >> 4;     // 0/1
    const int l16  = lane & 15;

    v8f acc[2][4] = {};

    const int kTiles = (K + BK - 1) / BK;
    for (int kt = 0; kt < kTiles; ++kt) {
        const int k0 = kt * BK;

        // ---- stage A tile (BM x BK) fp32 -> f16 in LDS ----
        {
            int r  = tid >> 3;           // 0..31
            int c4 = (tid & 7) * 4;      // 0..28 step 4
            #pragma unroll
            for (int i = 0; i < 4; ++i, r += 32) {
                int gm = m0 + r;
                int gk = k0 + c4;
                float4 vld;
                if (FULL) {
                    vld = *(const float4*)(A + (size_t)gm * K + gk);
                    // speculative prefetch of the next K tile (dropped if OOB)
                    __builtin_prefetch(A + (size_t)gm * K + gk + BK, 0, 1);
                } else {
                    vld = make_float4(0.f, 0.f, 0.f, 0.f);
                    if (gm < M) {
                        if (gk + 3 < K) {
                            vld = *(const float4*)(A + (size_t)gm * K + gk);
                        } else {
                            float e0 = (gk + 0 < K) ? A[(size_t)gm * K + gk + 0] : 0.f;
                            float e1 = (gk + 1 < K) ? A[(size_t)gm * K + gk + 1] : 0.f;
                            float e2 = (gk + 2 < K) ? A[(size_t)gm * K + gk + 2] : 0.f;
                            float e3 = (gk + 3 < K) ? A[(size_t)gm * K + gk + 3] : 0.f;
                            vld = make_float4(e0, e1, e2, e3);
                        }
                    }
                }
                _Float16* dst = &lA[r * BK + c4];
                dst[0] = (_Float16)vld.x; dst[1] = (_Float16)vld.y;
                dst[2] = (_Float16)vld.z; dst[3] = (_Float16)vld.w;
            }
        }
        // ---- stage B tile (BK x BN), transposed into lB[n][k] ----
        // each thread packs 2 consecutive k values -> one 32-bit DS store
        {
            int n   = tid & 127;
            int kb2 = (tid >> 7) * 2;    // 0 or 2
            #pragma unroll
            for (int i = 0; i < 8; ++i) {
                int kk  = kb2 + 4 * i;   // even; covers 0..31 over both halves
                int gk0 = k0 + kk;
                float v0, v1;
                if (FULL) {
                    v0 = W[(size_t)gk0 * N + (n0 + n)];
                    v1 = W[(size_t)(gk0 + 1) * N + (n0 + n)];
                } else {
                    v0 = (gk0 + 0 < K) ? W[(size_t)(gk0 + 0) * N + (n0 + n)] : 0.f;
                    v1 = (gk0 + 1 < K) ? W[(size_t)(gk0 + 1) * N + (n0 + n)] : 0.f;
                }
                union { _Float16 h[2]; unsigned int u; } pk;
                pk.h[0] = (_Float16)v0;
                pk.h[1] = (_Float16)v1;
                *reinterpret_cast<unsigned int*>(&lB[n * BK + kk]) = pk.u;
            }
        }
        __syncthreads();

        // ---- build fragments per ISA 7.12.2 f16 layouts ----
        v16h afr[2], bfr[4];
        {
            const int kb = half * 8;
            #pragma unroll
            for (int ti = 0; ti < 2; ++ti) {
                const _Float16* pr = &lA[(wm * 32 + ti * 16 + l16) * BK];
                #pragma unroll
                for (int j = 0; j < 8; ++j) {
                    afr[ti][j]     = pr[kb + j];
                    afr[ti][8 + j] = pr[16 + kb + j];
                }
            }
            #pragma unroll
            for (int tj = 0; tj < 4; ++tj) {
                const _Float16* pc = &lB[(wn * 64 + tj * 16 + l16) * BK + half * 8];
                #pragma unroll
                for (int j = 0; j < 8; ++j) {
                    bfr[tj][j]     = pc[j];
                    bfr[tj][8 + j] = pc[16 + j];
                }
            }
        }
        #pragma unroll
        for (int ti = 0; ti < 2; ++ti)
            #pragma unroll
            for (int tj = 0; tj < 4; ++tj)
                acc[ti][tj] = __builtin_amdgcn_wmma_f32_16x16x32_f16(
                    false, afr[ti], false, bfr[tj], (short)0, acc[ti][tj],
                    false, false);
        __syncthreads();
    }

    // ---- store D (16x16 f32 layout: lane 0-15 N=lane,M=r; 16-31 N=lane-16,M=r+8)
    #pragma unroll
    for (int ti = 0; ti < 2; ++ti) {
        #pragma unroll
        for (int tj = 0; tj < 4; ++tj) {
            int nCol = n0 + wn * 64 + tj * 16 + l16;
            float bv = bias ? bias[nCol] : 0.f;
            #pragma unroll
            for (int r = 0; r < 8; ++r) {
                int gm = m0 + wm * 32 + ti * 16 + r + half * 8;
                if (FULL || gm < M)
                    C[(size_t)gm * N + nCol] = acc[ti][tj][r] + bv;
            }
        }
    }
}

// ---------------------------------------------------------------------------
// helpers
// ---------------------------------------------------------------------------
__device__ __forceinline__ float block_reduce_sum(float v, float* sh)
{
    int tid = threadIdx.x;
    sh[tid] = v;
    __syncthreads();
    for (int off = blockDim.x >> 1; off > 0; off >>= 1) {
        if (tid < off) sh[tid] += sh[tid + off];
        __syncthreads();
    }
    float r = sh[0];
    __syncthreads();
    return r;
}

// y = elu((x-mu)/sqrt(var+eps)*g + b) [+ res]; one block per row; in-place safe
__global__ __launch_bounds__(256)
void ln_elu_kernel(const float* __restrict__ in, const float* __restrict__ g,
                   const float* __restrict__ b, const float* __restrict__ res,
                   float* __restrict__ out, int N)
{
    __shared__ float red[256];
    const size_t row = blockIdx.x;
    const float* x = in + row * N;
    float s = 0.f;
    for (int i = threadIdx.x; i < N; i += 256) s += x[i];
    float mu = block_reduce_sum(s, red) / (float)N;
    float v2 = 0.f;
    for (int i = threadIdx.x; i < N; i += 256) { float d = x[i] - mu; v2 += d * d; }
    float rstd = rsqrtf(block_reduce_sum(v2, red) / (float)N + 1e-5f);
    float* o = out + row * N;
    const float* rr = res ? res + row * N : nullptr;
    for (int i = threadIdx.x; i < N; i += 256) {
        float y = (x[i] - mu) * rstd * g[i] + b[i];
        y = y > 0.f ? y : (expf(y) - 1.f);
        if (rr) y += rr[i];
        o[i] = y;
    }
}

// masked axis-LN over (T,P) per (b,a) -> h0[..., c=p]
__global__ __launch_bounds__(256)
void feat_nx_kernel(const float* __restrict__ x, const float* __restrict__ g,
                    const float* __restrict__ bb, float* __restrict__ h0)
{
    __shared__ float red[256];
    const int b = blockIdx.x / 3, a = blockIdx.x % 3;
    float s0 = 0.f, s1 = 0.f, s2 = 0.f;
    for (int i = threadIdx.x; i < TT * 104; i += 256) {
        int t = i / 104, p = i % 104;
        float v = x[((size_t)(b * TT + t) * 104 + p) * 3 + a];
        float w = (v != 0.f) ? 1.f : 0.f;
        s0 += w; s1 += v * w; s2 += v * v * w;
    }
    s0 = block_reduce_sum(s0, red);
    s1 = block_reduce_sum(s1, red);
    s2 = block_reduce_sum(s2, red);
    float d = s0 + 1e-5f;
    float mu = s1 / d;
    float e = (s2 - 2.f * mu * s1 + mu * mu * s0) / d;
    float sd = sqrtf(e > 0.f ? e : 0.f);
    for (int i = threadIdx.x; i < TT * 104; i += 256) {
        int t = i / 104, p = i % 104;
        float v = x[((size_t)(b * TT + t) * 104 + p) * 3 + a];
        float w = (v != 0.f) ? 1.f : 0.f;
        float y = (g[p * 3 + a] * ((v - mu) / (sd + 1e-5f)) + bb[p * 3 + a]) * w;
        h0[(size_t)(b * TT + t) * 624 + p * 3 + a] = y;
    }
}

// per-(b,t,a,range) masked LN over P-slice -> h0[..., c=104+p]
__global__ __launch_bounds__(32)
void feat_rng_kernel(const float* __restrict__ x,
                     const float* g0, const float* b0, const float* g1, const float* b1,
                     const float* g2, const float* b2, const float* g3, const float* b3,
                     float* __restrict__ h0)
{
    const int bt = blockIdx.x;
    const int tid = threadIdx.x;
    if (tid >= 12) return;
    const int rid = tid / 3, a = tid % 3;
    const int S[4] = {0, 40, 62, 83}, E[4] = {40, 62, 83, 104};
    const float* gs[4] = {g0, g1, g2, g3};
    const float* bs[4] = {b0, b1, b2, b3};
    const int s = S[rid], e = E[rid];
    const float* xr = x + (size_t)bt * 104 * 3;
    float s0 = 0.f, s1 = 0.f, s2 = 0.f;
    for (int p = s; p < e; ++p) {
        float v = xr[p * 3 + a];
        float w = (v != 0.f) ? 1.f : 0.f;
        s0 += w; s1 += v * w; s2 += v * v * w;
    }
    float d = s0 + 1e-5f, mu = s1 / d;
    float ee = (s2 - 2.f * mu * s1 + mu * mu * s0) / d;
    float sd = sqrtf(ee > 0.f ? ee : 0.f);
    for (int p = s; p < e; ++p) {
        float v = xr[p * 3 + a];
        float w = (v != 0.f) ? 1.f : 0.f;
        float y = (gs[rid][(p - s) * 3 + a] * ((v - mu) / (sd + 1e-5f)) +
                   bs[rid][(p - s) * 3 + a]) * w;
        h0[(size_t)bt * 624 + (104 + p) * 3 + a] = y;
    }
}

__global__ __launch_bounds__(256)
void posenc_kernel(float* __restrict__ h, int total)
{
    int i = blockIdx.x * 256 + threadIdx.x;
    if (i >= total) return;
    int d = i % DIMM;
    int t = (i / DIMM) % TT;
    int i2 = d & ~1;
    float dv = expf((float)i2 * (-9.210340371976184f / (float)DIMM));
    float ang = (float)t * dv;
    h[i] += (d & 1) ? cosf(ang) : sinf(ang);
}

__global__ __launch_bounds__(256)
void exp_kernel(const float* __restrict__ in, float* __restrict__ out, int total,
                const unsigned char* __restrict__ mask, int rowdiv)
{
    int i = blockIdx.x * 256 + threadIdx.x;
    if (i >= total) return;
    float valid = 1.f;
    if (mask) valid = mask[i / rowdiv] ? 0.f : 1.f;
    out[i] = expf(in[i]) * valid;
}

__global__ __launch_bounds__(128)
void colsum_kernel(const float* __restrict__ ae, float* __restrict__ denom)
{
    int b = blockIdx.x, h = threadIdx.x;
    float s = 0.f;
    for (int t = 0; t < TT; ++t) s += ae[((size_t)b * TT + t) * HH + h];
    denom[b * HH + h] = s;
}

__global__ __launch_bounds__(256)
void div_kernel(const float* __restrict__ ae, const float* __restrict__ denom,
                float* __restrict__ attn, int total)
{
    int i = blockIdx.x * 256 + threadIdx.x;
    if (i >= total) return;
    int h = i % HH;
    int b = i / (TT * HH);
    attn[i] = ae[i] / (denom[b * HH + h] + 1e-5f);
}

__global__ __launch_bounds__(896)
void gpool_kernel(const float* __restrict__ attn, const float* __restrict__ v,
                  float* __restrict__ gp)
{
    int b = blockIdx.x, c = threadIdx.x;   // c = h*7+d
    int h = c / HD;
    float acc = 0.f;
    for (int t = 0; t < TT; ++t)
        acc += attn[((size_t)b * TT + t) * HH + h] *
               v[((size_t)b * TT + t) * DIMM + c];
    gp[b * DIMM + c] = acc;
}

__global__ __launch_bounds__(256)
void window_kernel(const float* __restrict__ attn_exp, const float* __restrict__ v,
                   const float* __restrict__ posw, const float* __restrict__ gp,
                   float* __restrict__ outw, int dil)
{
    int gid = blockIdx.x * 256 + threadIdx.x;
    if (gid >= BB * TT * HH) return;
    int h = gid % HH;
    int t = (gid / HH) % TT;
    int b = gid / (HH * TT);
    const float* pp = posw + b * 6 * HH;
    float aw[6];
    int src[5];
    aw[0] = pp[h];                       // aw==1 for global slot, times pos[0]
    float den = aw[0];
    #pragma unroll
    for (int j = 0; j < 5; ++j) {
        int s = t + dil * (j - 2);
        float ae = 0.f;
        if (s >= 0 && s < TT) ae = attn_exp[((size_t)b * TT + s) * HH + h];
        src[j] = s;
        aw[j + 1] = ae * pp[(j + 1) * HH + h];
        den += aw[j + 1];
    }
    float inv = 1.f / (den + 1e-5f);
    float o[HD];
    #pragma unroll
    for (int d = 0; d < HD; ++d) o[d] = aw[0] * inv * gp[b * DIMM + h * HD + d];
    #pragma unroll
    for (int j = 0; j < 5; ++j) {
        if (src[j] >= 0 && src[j] < TT) {
            float w = aw[j + 1] * inv;
            const float* vv = v + ((size_t)b * TT + src[j]) * DIMM + h * HD;
            #pragma unroll
            for (int d = 0; d < HD; ++d) o[d] += w * vv[d];
        }
    }
    float* op = outw + ((size_t)b * TT + t) * DIMM + h * HD;
    #pragma unroll
    for (int d = 0; d < HD; ++d) op[d] = o[d];
}

// ---------------------------------------------------------------------------
// host
// ---------------------------------------------------------------------------
static inline void gemm(const float* A, const float* W, const float* bias, float* C,
                        int M, int N, int K, hipStream_t s)
{
    dim3 g(N / BN, (M + BM - 1) / BM);
    const bool full = (M % BM == 0) && (K % BK == 0);
    if (full)
        wmma_gemm_kernel<true><<<g, dim3(256), 0, s>>>(A, W, bias, C, M, N, K);
    else
        wmma_gemm_kernel<false><<<g, dim3(256), 0, s>>>(A, W, bias, C, M, N, K);
}

extern "C" void kernel_launch(void* const* d_in, const int* in_sizes, int n_in,
                              void* d_out, int out_size, void* d_ws, size_t ws_size,
                              hipStream_t stream)
{
    (void)in_sizes; (void)n_in; (void)out_size; (void)ws_size;
    // input order: x, mask, then params in setup_inputs() insertion order
    const float* x = (const float*)d_in[0];
    const unsigned char* mask = (const unsigned char*)d_in[1];
    const float* P[120];
    for (int i = 2; i < 120 && i < n_in; ++i) P[i] = (const float*)d_in[i];

    // ---- workspace bump allocator (~233 MB) ----
    char* ws = (char*)d_ws;
    size_t off = 0;
    auto alloc = [&](size_t bytes) -> float* {
        float* p = (float*)(ws + off);
        off += (bytes + 255) & ~(size_t)255;
        return p;
    };
    const size_t M = BT_TOTAL;
    float* buf0 = alloc(M * 1792 * 4);   // widest scratch (in_w1 out, out-proj out)
    float* h    = alloc(M * DIMM * 4);   // hidden state
    float* t896 = alloc(M * DIMM * 4);   // pre-LN / window-out scratch
    float* vbuf = alloc(M * DIMM * 4);   // v;  also aliases h0 (624-wide) early
    float* ae   = alloc(M * HH * 4);     // attn scores -> attn_exp (in place)
    float* at   = alloc(M * HH * 4);     // normalized attn
    float* den  = alloc(BB * HH * 4);
    float* gp   = alloc(BB * DIMM * 4);
    float* h1   = alloc(BB * HH * 4);
    float* pw   = alloc(BB * 6 * HH * 4);
    float* h0   = vbuf;                  // h0 dead before v is first written

    // ---- features ----
    feat_nx_kernel<<<96, 256, 0, stream>>>(x, P[2], P[3], h0);
    feat_rng_kernel<<<M, 32, 0, stream>>>(x, P[4], P[5], P[6], P[7],
                                          P[8], P[9], P[10], P[11], h0);
    // ---- input projections ----
    gemm(h0, P[12], P[13], buf0, M, 1792, 624, stream);
    ln_elu_kernel<<<M, 256, 0, stream>>>(buf0, P[14], P[15], nullptr, buf0, 1792);
    gemm(buf0, P[16], P[17], t896, M, DIMM, 1792, stream);
    ln_elu_kernel<<<M, 256, 0, stream>>>(t896, P[18], P[19], nullptr, h, DIMM);
    posenc_kernel<<<(int)((M * DIMM + 255) / 256), 256, 0, stream>>>(h, (int)(M * DIMM));

    const int dils[6] = {1, 3, 3, 3, 3, 3};
    for (int L = 0; L < 6; ++L) {
        const int pb = 20 + 16 * L;
        // attention logits -> exp*valid -> softmax-over-time
        gemm(h, P[pb + 0], P[pb + 1], ae, M, HH, DIMM, stream);
        exp_kernel<<<(int)((M * HH + 255) / 256), 256, 0, stream>>>(
            ae, ae, (int)(M * HH), mask, HH);
        colsum_kernel<<<BB, HH, 0, stream>>>(ae, den);
        div_kernel<<<(int)((M * HH + 255) / 256), 256, 0, stream>>>(
            ae, den, at, (int)(M * HH));
        // v = elu(ln(x @ v_w + v_b))
        gemm(h, P[pb + 8], P[pb + 9], t896, M, DIMM, DIMM, stream);
        ln_elu_kernel<<<M, 256, 0, stream>>>(t896, P[pb + 10], P[pb + 11],
                                             nullptr, vbuf, DIMM);
        // global pool + position MLP
        gpool_kernel<<<BB, DIMM, 0, stream>>>(at, vbuf, gp);
        gemm(gp, P[pb + 2], P[pb + 3], h1, BB, HH, DIMM, stream);
        ln_elu_kernel<<<BB, 256, 0, stream>>>(h1, P[pb + 4], P[pb + 5],
                                              nullptr, h1, HH);
        gemm(h1, P[pb + 6], P[pb + 7], pw, BB, 6 * HH, HH, stream);
        exp_kernel<<<(BB * 6 * HH + 255) / 256, 256, 0, stream>>>(
            pw, pw, BB * 6 * HH, nullptr, 1);
        // dilated window combine
        window_kernel<<<(int)((BB * TT * HH + 255) / 256), 256, 0, stream>>>(
            ae, vbuf, pw, gp, t896, dils[L]);
        // output projection + LN + ELU + residual (written back into h)
        gemm(t896, P[pb + 12], P[pb + 13], buf0, M, DIMM, DIMM, stream);
        ln_elu_kernel<<<M, 256, 0, stream>>>(buf0, P[pb + 14], P[pb + 15], h, h, DIMM);
    }

    // ---- final projection directly into d_out, LN+ELU in place ----
    float* out = (float*)d_out;
    gemm(h, P[116], P[117], out, M, 2048, DIMM, stream);
    ln_elu_kernel<<<M, 256, 0, stream>>>(out, P[118], P[119], nullptr, out, 2048);
}
